// NDLMOEDecoderlayer_51788715655578
// MI455X (gfx1250) — compile-verified, matrix operand order checked by reading, not run on hardware
//
#include <hip/hip_runtime.h>
#include <hip/hip_bf16.h>

// ---------------- types for WMMA operands ----------------
typedef __attribute__((ext_vector_type(16))) __bf16        v16bf;
typedef __attribute__((ext_vector_type(8)))  float         v8f;
typedef __attribute__((ext_vector_type(8)))  unsigned short u16x8;
typedef int v4i __attribute__((vector_size(16)));   // matches async-LDS builtin pointee

union Frag16 {
  v16bf v;
  u16x8 h[2];
};

__device__ __forceinline__ unsigned short f2bf(float f) {
  unsigned int u = __float_as_uint(f);
  unsigned int r = u + 0x7FFFu + ((u >> 16) & 1u);   // round-to-nearest-even
  return (unsigned short)(r >> 16);
}

__device__ __forceinline__ v8f wmma_bf16(const Frag16& a, const Frag16& b, v8f c) {
  return __builtin_amdgcn_wmma_f32_16x16x32_bf16(false, a.v, false, b.v, (short)0, c, false, false);
}

#if __has_builtin(__builtin_amdgcn_global_load_async_to_lds_b128) && \
    __has_builtin(__builtin_amdgcn_s_wait_asynccnt)
#define HAVE_ASYNC_LDS 1
#else
#define HAVE_ASYNC_LDS 0
#endif

// 16-byte global -> LDS copy (async DMA path on CDNA5, register relay otherwise)
__device__ __forceinline__ void cp_g2l_b128(const unsigned short* g, unsigned short* l) {
#if HAVE_ASYNC_LDS
  __builtin_amdgcn_global_load_async_to_lds_b128(
      (__attribute__((address_space(1))) v4i*)(g),
      (__attribute__((address_space(3))) v4i*)(l), 0, 0);
#else
  *(u16x8*)l = *(const u16x8*)g;
#endif
}

__device__ __forceinline__ void cp_wait_all() {
#if HAVE_ASYNC_LDS
  __builtin_amdgcn_s_wait_asynccnt(0);
#endif
}

#define SS 1024
#define DD 1024
#define NH 16
#define NKVH 8
#define HDIM 64
#define NE 8
#define EINTER 512
#define SHINTER 1024
#define NT 1024

// ---------------- fp32 -> bf16 convert (vectorized x4) ----------------
__global__ void cvt_bf16_kernel(const float* __restrict__ s, unsigned short* __restrict__ d, int n) {
  int i4 = blockIdx.x * blockDim.x + threadIdx.x;
  int base = i4 * 4;
  if (base >= n) return;
  float4 v = ((const float4*)s)[i4];
  ushort4 o;
  o.x = f2bf(v.x); o.y = f2bf(v.y); o.z = f2bf(v.z); o.w = f2bf(v.w);
  ((ushort4*)d)[i4] = o;
}

// ---------------- RMSNorm (one block per token) ----------------
__global__ void __launch_bounds__(256) rmsnorm_kernel(const float* __restrict__ x,
                                                      const float* __restrict__ w,
                                                      unsigned short* __restrict__ obf,
                                                      float* __restrict__ of, int D) {
  int row = blockIdx.x;
  const float4* xr = (const float4*)(x + (long long)row * D);
  float4 xv = xr[threadIdx.x];
  float ss = xv.x * xv.x + xv.y * xv.y + xv.z * xv.z + xv.w * xv.w;
#pragma unroll
  for (int off = 16; off > 0; off >>= 1) ss += __shfl_xor(ss, off, 32);
  __shared__ float red[8];
  if ((threadIdx.x & 31) == 0) red[threadIdx.x >> 5] = ss;
  __syncthreads();
  float tot = 0.f;
#pragma unroll
  for (int i = 0; i < 8; ++i) tot += red[i];
  float sc = rsqrtf(tot / (float)D + 1e-6f);
  const float4* wr = (const float4*)w;
  float4 wv = wr[threadIdx.x];
  float o0 = xv.x * sc * wv.x, o1 = xv.y * sc * wv.y;
  float o2 = xv.z * sc * wv.z, o3 = xv.w * sc * wv.w;
  long long base = (long long)row * D + threadIdx.x * 4;
  obf[base + 0] = f2bf(o0); obf[base + 1] = f2bf(o1);
  obf[base + 2] = f2bf(o2); obf[base + 3] = f2bf(o3);
  if (of) { of[base + 0] = o0; of[base + 1] = o1; of[base + 2] = o2; of[base + 3] = o3; }
}

// ---------------- C = A(MxK) * B(NxK)^T via WMMA bf16, LDS-staged, double-buffered ------
// EPI: 0 = store f32, 1 = store f32 + residual, 2 = atomicAdd(C, scale[row]*acc)
template <int EPI>
__global__ void __launch_bounds__(256) gemm_nt_kernel(
    const unsigned short* __restrict__ A, const unsigned short* __restrict__ B,
    float* __restrict__ C, const float* __restrict__ R,
    const float* __restrict__ scale, int scaleStride,
    int M, int N, int K,
    long long aStride, long long bStride, long long cStride) {
  __shared__ unsigned short aTile[2][64 * 32];    //  8 KB
  __shared__ unsigned short bTile[2][128 * 32];   // 16 KB
  int z = blockIdx.z;
  A += (long long)z * aStride;
  B += (long long)z * bStride;
  C += (long long)z * cStride;
  const float* sc = scale ? (scale + z) : nullptr;

  int tid  = threadIdx.x;
  int lane = tid & 31;
  int wave = tid >> 5;
  int half = lane >> 4;
  int r    = lane & 15;
  int wm = wave & 3, wn = wave >> 2;
  int rowBlk = blockIdx.x * 64;
  int colBlk = blockIdx.y * 128;

  // copy mapping: chunk c covers 8 contiguous K elems; tid*8 linearizes [row][k]
  const unsigned short* aSrc  = A + (long long)(rowBlk + (tid >> 2)) * K + (tid & 3) * 8;
  const unsigned short* bSrc0 = B + (long long)(colBlk + (tid >> 2)) * K + (tid & 3) * 8;
  const unsigned short* bSrc1 = B + (long long)(colBlk + 64 + (tid >> 2)) * K + (tid & 3) * 8;

  v8f acc[4];
#pragma unroll
  for (int t = 0; t < 4; ++t)
#pragma unroll
    for (int i = 0; i < 8; ++i) acc[t][i] = 0.f;

  auto fill = [&](int buf, int k0) {
    cp_g2l_b128(aSrc + k0,  &aTile[buf][tid * 8]);
    cp_g2l_b128(bSrc0 + k0, &bTile[buf][tid * 8]);
    cp_g2l_b128(bSrc1 + k0, &bTile[buf][tid * 8 + 2048]);
  };

  int nk = K >> 5;
  fill(0, 0);
  for (int ks = 0; ks < nk; ++ks) {
    cp_wait_all();
    __syncthreads();
    if (ks + 1 < nk) fill((ks + 1) & 1, (ks + 1) * 32);  // DMA overlaps WMMAs below
    int buf = ks & 1;
    // issue ALL fragment loads first (one DS clause), then the WMMA chain
    Frag16 af, bfr[4];
    const unsigned short* ap = &aTile[buf][(wm * 16 + r) * 32 + half * 8];
    af.h[0] = *(const u16x8*)(ap);
    af.h[1] = *(const u16x8*)(ap + 16);
#pragma unroll
    for (int t = 0; t < 4; ++t) {
      const unsigned short* bp = &bTile[buf][(wn * 64 + t * 16 + r) * 32 + half * 8];
      bfr[t].h[0] = *(const u16x8*)(bp);
      bfr[t].h[1] = *(const u16x8*)(bp + 16);
    }
#pragma unroll
    for (int t = 0; t < 4; ++t) acc[t] = wmma_bf16(af, bfr[t], acc[t]);
    __syncthreads();
  }

#pragma unroll
  for (int t = 0; t < 4; ++t) {
    int col = colBlk + wn * 64 + t * 16 + r;
#pragma unroll
    for (int i = 0; i < 8; ++i) {
      int row = rowBlk + wm * 16 + i + half * 8;
      long long idx = (long long)row * N + col;
      float v = acc[t][i];
      if (EPI == 0) {
        C[idx] = v;
      } else if (EPI == 1) {
        C[idx] = v + R[idx];
      } else {
        atomicAdd(&C[idx], sc[(long long)row * scaleStride] * v);
      }
    }
  }
}

// ------ fused dual-GEMM + SwiGLU: G = (A*B1^T) * silu(A*B2^T), bf16 out, LDS-staged ------
__global__ void __launch_bounds__(256) gemm_swiglu_kernel(
    const unsigned short* __restrict__ A, const unsigned short* __restrict__ B1,
    const unsigned short* __restrict__ B2, unsigned short* __restrict__ G,
    int M, int N, int K, long long bStride, long long gStride) {
  __shared__ unsigned short aTile[2][64 * 32];
  __shared__ unsigned short b1Tile[2][128 * 32];
  __shared__ unsigned short b2Tile[2][128 * 32];
  int z = blockIdx.z;
  B1 += (long long)z * bStride;
  B2 += (long long)z * bStride;
  G  += (long long)z * gStride;

  int tid  = threadIdx.x;
  int lane = tid & 31;
  int wave = tid >> 5;
  int half = lane >> 4;
  int r    = lane & 15;
  int wm = wave & 3, wn = wave >> 2;
  int rowBlk = blockIdx.x * 64;
  int colBlk = blockIdx.y * 128;

  long long bOff0 = (long long)(colBlk + (tid >> 2)) * K + (tid & 3) * 8;
  long long bOff1 = (long long)(colBlk + 64 + (tid >> 2)) * K + (tid & 3) * 8;
  const unsigned short* aSrc = A + (long long)(rowBlk + (tid >> 2)) * K + (tid & 3) * 8;

  v8f acc1[4], acc2[4];
#pragma unroll
  for (int t = 0; t < 4; ++t)
#pragma unroll
    for (int i = 0; i < 8; ++i) { acc1[t][i] = 0.f; acc2[t][i] = 0.f; }

  auto fill = [&](int buf, int k0) {
    cp_g2l_b128(aSrc + k0,        &aTile[buf][tid * 8]);
    cp_g2l_b128(B1 + bOff0 + k0,  &b1Tile[buf][tid * 8]);
    cp_g2l_b128(B1 + bOff1 + k0,  &b1Tile[buf][tid * 8 + 2048]);
    cp_g2l_b128(B2 + bOff0 + k0,  &b2Tile[buf][tid * 8]);
    cp_g2l_b128(B2 + bOff1 + k0,  &b2Tile[buf][tid * 8 + 2048]);
  };

  int nk = K >> 5;
  fill(0, 0);
  for (int ks = 0; ks < nk; ++ks) {
    cp_wait_all();
    __syncthreads();
    if (ks + 1 < nk) fill((ks + 1) & 1, (ks + 1) * 32);
    int buf = ks & 1;
    Frag16 af, b1f[4], b2f[4];
    const unsigned short* ap = &aTile[buf][(wm * 16 + r) * 32 + half * 8];
    af.h[0] = *(const u16x8*)(ap);
    af.h[1] = *(const u16x8*)(ap + 16);
#pragma unroll
    for (int t = 0; t < 4; ++t) {
      int cl = (wn * 64 + t * 16 + r) * 32 + half * 8;
      b1f[t].h[0] = *(const u16x8*)(&b1Tile[buf][cl]);
      b1f[t].h[1] = *(const u16x8*)(&b1Tile[buf][cl] + 16);
      b2f[t].h[0] = *(const u16x8*)(&b2Tile[buf][cl]);
      b2f[t].h[1] = *(const u16x8*)(&b2Tile[buf][cl] + 16);
    }
#pragma unroll
    for (int t = 0; t < 4; ++t) {
      acc1[t] = wmma_bf16(af, b1f[t], acc1[t]);
      acc2[t] = wmma_bf16(af, b2f[t], acc2[t]);
    }
    __syncthreads();
  }

#pragma unroll
  for (int t = 0; t < 4; ++t) {
    int col = colBlk + wn * 64 + t * 16 + r;
#pragma unroll
    for (int i = 0; i < 8; ++i) {
      int row = rowBlk + wm * 16 + i + half * 8;
      float a1 = acc1[t][i], a2 = acc2[t][i];
      float g = a1 * (a2 / (1.0f + __expf(-a2)));
      G[(long long)row * N + col] = f2bf(g);
    }
  }
}

// ---------------- RoPE (f32 in, bf16 out) ----------------
__global__ void rope_kernel(const float* __restrict__ in, unsigned short* __restrict__ out,
                            const int* __restrict__ pos, int S, int nheads) {
  int idx = blockIdx.x * blockDim.x + threadIdx.x;
  int total = S * nheads * HDIM;
  if (idx >= total) return;
  int d = idx & 63;
  int s = idx / (nheads * HDIM);
  int p = pos[s];
  int i = d & 31;
  float freq = __powf(10000.0f, -(float)i / 32.0f);
  float th = (float)p * freq;
  float c = __cosf(th), sn = __sinf(th);
  int base = idx - d;
  float x  = in[base + d];
  float xr = (d < 32) ? -in[base + d + 32] : in[base + d - 32];
  out[idx] = f2bf(x * c + xr * sn);
}

// ---------------- V transpose: [S, KVH*HD] f32 -> [KVH*HD, S] bf16 ----------------
__global__ void vtrans_kernel(const float* __restrict__ v, unsigned short* __restrict__ vt,
                              int S, int C) {
  int idx = blockIdx.x * blockDim.x + threadIdx.x;
  if (idx >= S * C) return;
  int c = idx % C;
  int s = idx / C;
  vt[(long long)c * S + s] = f2bf(v[idx]);
}

// ---------------- flash attention: one wave = (head, 16-query tile) ----------------
__global__ void __launch_bounds__(256) attn_kernel(
    const unsigned short* __restrict__ qb,  // [S, NH*HD] bf16 (rope'd)
    const unsigned short* __restrict__ kb,  // [S, NKVH*HD] bf16 (rope'd)
    const unsigned short* __restrict__ vt,  // [NKVH*HD, S] bf16
    unsigned short* __restrict__ ob,        // [S, NH*HD] bf16
    int S) {
  __shared__ unsigned short lds_p[8][16 * 32];
  int wave = threadIdx.x >> 5;
  int lane = threadIdx.x & 31;
  int half = lane >> 4;
  int r    = lane & 15;
  int task = blockIdx.x * 8 + wave;
  int head = task >> 6;  // 64 m-tiles per head
  int mt   = task & 63;
  int kvh  = head >> 1;  // repeat_kv: H/KVH == 2

  Frag16 qf[2];
  {
    const unsigned short* qp = qb + (long long)(mt * 16 + r) * (NH * HDIM) + head * HDIM;
#pragma unroll
    for (int j = 0; j < 2; ++j) {
      qf[j].h[0] = *(const u16x8*)(qp + j * 32 + half * 8);
      qf[j].h[1] = *(const u16x8*)(qp + j * 32 + half * 8 + 16);
    }
  }

  float mrun[8], lrun[8];
  v8f oacc[4];
#pragma unroll
  for (int i = 0; i < 8; ++i) { mrun[i] = -1e30f; lrun[i] = 0.f; }
#pragma unroll
  for (int t = 0; t < 4; ++t)
#pragma unroll
    for (int i = 0; i < 8; ++i) oacc[t][i] = 0.f;

  int nchunks = mt / 2 + 1;
  for (int c2 = 0; c2 < nchunks; ++c2) {
    int kbase = c2 * 32;
    float s[2][8];
#pragma unroll
    for (int jj = 0; jj < 2; ++jj) {
      int kb0 = kbase + jj * 16;
      if (kb0 < (mt + 1) * 16) {
        v8f sacc;
#pragma unroll
        for (int i = 0; i < 8; ++i) sacc[i] = 0.f;
        Frag16 kf[2];
#pragma unroll
        for (int j = 0; j < 2; ++j) {
          const unsigned short* kp =
              kb + (long long)(kb0 + r) * (NKVH * HDIM) + kvh * HDIM + j * 32 + half * 8;
          kf[j].h[0] = *(const u16x8*)(kp);
          kf[j].h[1] = *(const u16x8*)(kp + 16);
        }
#pragma unroll
        for (int j = 0; j < 2; ++j) sacc = wmma_bf16(qf[j], kf[j], sacc);
        int key = kb0 + r;
#pragma unroll
        for (int i = 0; i < 8; ++i) {
          int qrow = mt * 16 + i + half * 8;
          float sv = sacc[i] * 0.125f;  // 1/sqrt(64)
          s[jj][i] = (key <= qrow) ? sv : -1e30f;
        }
      } else {
#pragma unroll
        for (int i = 0; i < 8; ++i) s[jj][i] = -1e30f;
      }
    }

    float alpha[8];
#pragma unroll
    for (int i = 0; i < 8; ++i) {
      float mx = fmaxf(s[0][i], s[1][i]);
#pragma unroll
      for (int off = 1; off < 16; off <<= 1) mx = fmaxf(mx, __shfl_xor(mx, off, 32));
      float mnew = fmaxf(mrun[i], mx);
      alpha[i] = __expf(mrun[i] - mnew);
      mrun[i] = mnew;
      lrun[i] *= alpha[i];
    }
#pragma unroll
    for (int t = 0; t < 4; ++t)
#pragma unroll
      for (int i = 0; i < 8; ++i) oacc[t][i] *= alpha[i];

#pragma unroll
    for (int jj = 0; jj < 2; ++jj)
#pragma unroll
      for (int i = 0; i < 8; ++i) {
        float p = __expf(s[jj][i] - mrun[i]);
        lrun[i] += p;
        lds_p[wave][(i + half * 8) * 32 + jj * 16 + r] = f2bf(p);
      }

    // load P fragment and ALL V fragments first, then the WMMA chain
    Frag16 pf, vf[4];
    {
      const unsigned short* pp = &lds_p[wave][r * 32 + half * 8];
      pf.h[0] = *(const u16x8*)(pp);
      pf.h[1] = *(const u16x8*)(pp + 16);
    }
#pragma unroll
    for (int t = 0; t < 4; ++t) {
      const unsigned short* vp =
          vt + (long long)(kvh * HDIM + t * 16 + r) * S + kbase + half * 8;
      vf[t].h[0] = *(const u16x8*)(vp);
      vf[t].h[1] = *(const u16x8*)(vp + 16);
    }
#pragma unroll
    for (int t = 0; t < 4; ++t) oacc[t] = wmma_bf16(pf, vf[t], oacc[t]);
  }

  float linv[8];
#pragma unroll
  for (int i = 0; i < 8; ++i) {
    float ls = lrun[i];
#pragma unroll
    for (int off = 1; off < 16; off <<= 1) ls += __shfl_xor(ls, off, 32);
    linv[i] = 1.0f / ls;
  }
#pragma unroll
  for (int t = 0; t < 4; ++t)
#pragma unroll
    for (int i = 0; i < 8; ++i) {
      int qrow = mt * 16 + i + half * 8;
      int col = head * HDIM + t * 16 + r;
      ob[(long long)qrow * (NH * HDIM) + col] = f2bf(oacc[t][i] * linv[i]);
    }
}

// ---------------- router: softmax over 8 experts, top-2, renormalized ----------------
__global__ void __launch_bounds__(256) router_kernel(const float* __restrict__ xn,
                                                     const float* __restrict__ gw,
                                                     float* __restrict__ wfull, int T, int D) {
  int wave = threadIdx.x >> 5;
  int lane = threadIdx.x & 31;
  int t = blockIdx.x * 8 + wave;
  if (t >= T) return;
  float acc[NE];
#pragma unroll
  for (int e = 0; e < NE; ++e) acc[e] = 0.f;
  const float* xr = xn + (long long)t * D;
  for (int j = lane; j < D; j += 32) {
    float xv = xr[j];
#pragma unroll
    for (int e = 0; e < NE; ++e) acc[e] += xv * gw[e * D + j];
  }
#pragma unroll
  for (int e = 0; e < NE; ++e)
#pragma unroll
    for (int off = 16; off > 0; off >>= 1) acc[e] += __shfl_xor(acc[e], off, 32);

  float mx = acc[0];
#pragma unroll
  for (int e = 1; e < NE; ++e) mx = fmaxf(mx, acc[e]);
  float p[NE], sume = 0.f;
#pragma unroll
  for (int e = 0; e < NE; ++e) { p[e] = __expf(acc[e] - mx); sume += p[e]; }
#pragma unroll
  for (int e = 0; e < NE; ++e) p[e] /= sume;
  int i1 = 0;
#pragma unroll
  for (int e = 1; e < NE; ++e) if (p[e] > p[i1]) i1 = e;
  int i2 = (i1 == 0) ? 1 : 0;
#pragma unroll
  for (int e = 0; e < NE; ++e) if (e != i1 && p[e] > p[i2]) i2 = e;
  float wsum = p[i1] + p[i2] + 1e-20f;
  if (lane == 0) {
#pragma unroll
    for (int e = 0; e < NE; ++e) wfull[t * NE + e] = 0.f;
    wfull[t * NE + i1] = p[i1] / wsum;
    wfull[t * NE + i2] = p[i2] / wsum;
  }
}

// ---------------- out = a + b ----------------
__global__ void add_kernel(const float* __restrict__ a, const float* __restrict__ b,
                           float* __restrict__ o, int n) {
  int i = blockIdx.x * blockDim.x + threadIdx.x;
  if (i < n) o[i] = a[i] + b[i];
}

// ============================================================================
extern "C" void kernel_launch(void* const* d_in, const int* in_sizes, int n_in,
                              void* d_out, int out_size, void* d_ws, size_t ws_size,
                              hipStream_t stream) {
  (void)in_sizes; (void)n_in; (void)out_size; (void)ws_size;
  const float* hidden = (const float*)d_in[0];
  const int*   pos    = (const int*)d_in[1];
  const float* qw  = (const float*)d_in[2];
  const float* kw  = (const float*)d_in[3];
  const float* vw  = (const float*)d_in[4];
  const float* ow  = (const float*)d_in[5];
  const float* iln = (const float*)d_in[6];
  const float* pln = (const float*)d_in[7];
  const float* gatew = (const float*)d_in[8];
  const float* ew1 = (const float*)d_in[9];
  const float* ew2 = (const float*)d_in[10];
  const float* ewo = (const float*)d_in[11];
  const float* sw1 = (const float*)d_in[12];
  const float* sw2 = (const float*)d_in[13];
  const float* swo = (const float*)d_in[14];
  float* out = (float*)d_out;

  size_t off = 0;
  auto alloc = [&](size_t elems, size_t esz) -> void* {
    void* p = (char*)d_ws + off;
    off += ((elems * esz + 255) / 256) * 256;
    return p;
  };

  unsigned short* wq_b   = (unsigned short*)alloc(1024 * 1024, 2);
  unsigned short* wk_b   = (unsigned short*)alloc(512 * 1024, 2);
  unsigned short* wv_b   = (unsigned short*)alloc(512 * 1024, 2);
  unsigned short* wo_b   = (unsigned short*)alloc(1024 * 1024, 2);
  unsigned short* we1_b  = (unsigned short*)alloc((size_t)NE * EINTER * DD, 2);
  unsigned short* we2_b  = (unsigned short*)alloc((size_t)NE * EINTER * DD, 2);
  unsigned short* wewo_b = (unsigned short*)alloc((size_t)NE * DD * EINTER, 2);
  unsigned short* ws1_b  = (unsigned short*)alloc((size_t)SHINTER * DD, 2);
  unsigned short* ws2_b  = (unsigned short*)alloc((size_t)SHINTER * DD, 2);
  unsigned short* wswo_b = (unsigned short*)alloc((size_t)DD * SHINTER, 2);
  unsigned short* hn_b  = (unsigned short*)alloc((size_t)SS * DD, 2);
  float*          q_f   = (float*)alloc((size_t)SS * 1024, 4);
  float*          k_f   = (float*)alloc((size_t)SS * 512, 4);
  float*          v_f   = (float*)alloc((size_t)SS * 512, 4);
  unsigned short* q_b   = (unsigned short*)alloc((size_t)SS * 1024, 2);
  unsigned short* k_b   = (unsigned short*)alloc((size_t)SS * 512, 2);
  unsigned short* vt_b  = (unsigned short*)alloc((size_t)512 * SS, 2);
  unsigned short* ao_b  = (unsigned short*)alloc((size_t)SS * 1024, 2);
  float*          h2    = (float*)alloc((size_t)SS * DD, 4);
  unsigned short* xn_b  = (unsigned short*)alloc((size_t)SS * DD, 2);
  float*          xn_f  = (float*)alloc((size_t)SS * DD, 4);
  float*          wfull = (float*)alloc((size_t)NT * NE, 4);
  unsigned short* gs_b  = (unsigned short*)alloc((size_t)NT * SHINTER, 2);
  unsigned short* ge_b  = (unsigned short*)alloc((size_t)NE * NT * EINTER, 2);
  float*          y     = (float*)alloc((size_t)NT * DD, 4);

  auto cvt = [&](const float* s, unsigned short* d, int n) {
    cvt_bf16_kernel<<<n / 1024, 256, 0, stream>>>(s, d, n);
  };
  cvt(qw, wq_b, 1024 * 1024);
  cvt(kw, wk_b, 512 * 1024);
  cvt(vw, wv_b, 512 * 1024);
  cvt(ow, wo_b, 1024 * 1024);
  cvt(ew1, we1_b, NE * EINTER * DD);
  cvt(ew2, we2_b, NE * EINTER * DD);
  cvt(ewo, wewo_b, NE * DD * EINTER);
  cvt(sw1, ws1_b, SHINTER * DD);
  cvt(sw2, ws2_b, SHINTER * DD);
  cvt(swo, wswo_b, DD * SHINTER);

  rmsnorm_kernel<<<SS, 256, 0, stream>>>(hidden, iln, hn_b, nullptr, DD);

  gemm_nt_kernel<0><<<dim3(16, 8, 1), 256, 0, stream>>>(hn_b, wq_b, q_f, nullptr, nullptr, 0,
                                                        SS, 1024, DD, 0, 0, 0);
  gemm_nt_kernel<0><<<dim3(16, 4, 1), 256, 0, stream>>>(hn_b, wk_b, k_f, nullptr, nullptr, 0,
                                                        SS, 512, DD, 0, 0, 0);
  gemm_nt_kernel<0><<<dim3(16, 4, 1), 256, 0, stream>>>(hn_b, wv_b, v_f, nullptr, nullptr, 0,
                                                        SS, 512, DD, 0, 0, 0);

  rope_kernel<<<(SS * NH * HDIM) / 256, 256, 0, stream>>>(q_f, q_b, pos, SS, NH);
  rope_kernel<<<(SS * NKVH * HDIM) / 256, 256, 0, stream>>>(k_f, k_b, pos, SS, NKVH);
  vtrans_kernel<<<(SS * 512) / 256, 256, 0, stream>>>(v_f, vt_b, SS, 512);

  attn_kernel<<<128, 256, 0, stream>>>(q_b, k_b, vt_b, ao_b, SS);

  gemm_nt_kernel<1><<<dim3(16, 8, 1), 256, 0, stream>>>(ao_b, wo_b, h2, hidden, nullptr, 0,
                                                        SS, 1024, 1024, 0, 0, 0);

  rmsnorm_kernel<<<SS, 256, 0, stream>>>(h2, pln, xn_b, xn_f, DD);

  router_kernel<<<NT / 8, 256, 0, stream>>>(xn_f, gatew, wfull, NT, DD);

  gemm_swiglu_kernel<<<dim3(16, 8, 1), 256, 0, stream>>>(xn_b, ws1_b, ws2_b, gs_b,
                                                         NT, SHINTER, DD, 0, 0);
  gemm_nt_kernel<0><<<dim3(16, 8, 1), 256, 0, stream>>>(gs_b, wswo_b, y, nullptr, nullptr, 0,
                                                        NT, DD, SHINTER, 0, 0, 0);

  gemm_swiglu_kernel<<<dim3(16, 4, NE), 256, 0, stream>>>(
      xn_b, we1_b, we2_b, ge_b, NT, EINTER, DD,
      (long long)EINTER * DD, (long long)NT * EINTER);
  gemm_nt_kernel<2><<<dim3(16, 8, NE), 256, 0, stream>>>(
      ge_b, wewo_b, y, nullptr, wfull, NE,
      NT, DD, EINTER,
      (long long)NT * EINTER, (long long)DD * EINTER, 0);

  add_kernel<<<(SS * DD) / 256, 256, 0, stream>>>(h2, y, out, SS * DD);
}
